// GaussianVQ_1589137899971
// MI455X (gfx1250) — matrix operand order repaired and checked by reading, not compiled, not run on hardware
//
#include <hip/hip_runtime.h>
#include <hip/hip_fp16.h>

typedef __attribute__((ext_vector_type(16))) _Float16 v16h;
typedef __attribute__((ext_vector_type(8)))  _Float16 v8h;
typedef __attribute__((ext_vector_type(8)))  float    v8f;

#define NROWS     16        // rows per workgroup
#define NTOT      65536     // bs*W*H
#define NBLK      (NTOT / NROWS)   // 4096
#define XT_LD     72        // 64 + 8 f32 pad
#define ENC_LD    1032      // 1024 + 8 f16 pad -> 2064B row stride, conflict-free A loads
#define GL_LD     132       // gumbel LDS row stride (floats): 528B, 16B-aligned, bank-skewed
#define GL_WV     (16 * GL_LD)   // floats per wave chunk

// ---------------------------------------------------------------------------
// Pre-kernel: codebook f32 -> f16 (row major + transposed) + row norms ||c||^2
// ---------------------------------------------------------------------------
__global__ void vq_precompute(const float* __restrict__ cb,
                              _Float16* __restrict__ cb16,
                              _Float16* __restrict__ cbT16,
                              float* __restrict__ sqc) {
  int idx = blockIdx.x * blockDim.x + threadIdx.x;   // 0..65535
  int k = idx >> 6, d = idx & 63;
  float v = cb[idx];
  cb16[idx] = (_Float16)v;
  cbT16[d * 1024 + k] = (_Float16)v;
  if (d == 0) {
    float s = 0.f;
#pragma unroll
    for (int j = 0; j < 64; ++j) { float c = cb[k * 64 + j]; s += c * c; }
    sqc[k] = s;
  }
}

// ---------------------------------------------------------------------------
// Fused: async gumbel prefetch -> logits (WMMA) -> two softmaxes (flash
// combine) -> encodings (f16 LDS) -> quantized (WMMA) -> out + KL partials
// ---------------------------------------------------------------------------
__global__ __launch_bounds__(256)
void vq_fused(const float* __restrict__ x,
              const float* __restrict__ gum,
              const float* __restrict__ lvq,
              const _Float16* __restrict__ cb16,
              const _Float16* __restrict__ cbT16,
              const float* __restrict__ sqc,
              float* __restrict__ out_q,
              float* __restrict__ partials) {
  __shared__ __align__(16) float glds[8 * GL_WV];    // gumbel tile, per-wave chunks
  __shared__ float    xt[NROWS * XT_LD];             // x tile f32
  __shared__ _Float16 enc[NROWS * ENC_LD];           // encodings f16
  __shared__ float pmx[8][16], ps1[8][16], ps2[8][16], pgm[8][16], pgz[8][16];
  __shared__ float rowMg[16], rowIZ[16], kldrow[16];
  __shared__ float qpart[16][68];
  __shared__ float qfin[16][68];
  __shared__ float red[256];

  const int tid  = threadIdx.x;
  const int lane = tid & 31;
  const int wave = tid >> 5;
  const int ln   = lane & 15;      // column / row-within-half index
  const int h    = lane >> 4;      // half select (K/M split per ISA layouts)

  const int n0  = blockIdx.x * NROWS;
  const int b   = n0 >> 10;        // 16 | 1024 -> batch constant per WG
  const int nlo = n0 & 1023;
  const float w = 0.5f / (1.0f + __expf(lvq[0]));   // 0.5 * precision_q

  const int c0 = wave * 128;       // this wave's 128 codes

  // ---- async prefetch: this wave's gumbel chunk [16 rows][128 cols] -> LDS.
  // One b128 per lane per row: 512B/row, coalesced; tracked by ASYNCcnt.
  {
    const float* gwb = gum + (size_t)n0 * 1024 + c0 + lane * 4;
    unsigned lbase = (unsigned)(size_t)(&glds[wave * GL_WV]) + lane * 16;
#pragma unroll
    for (int i = 0; i < 16; ++i) {
      const float* ga = gwb + (size_t)i * 1024;
      unsigned la = lbase + i * (GL_LD * 4);
      asm volatile("global_load_async_to_lds_b128 %0, %1, off"
                   :: "v"(la), "v"(ga) : "memory");
    }
  }

  // ---- load x tile [16 rows][64 dims]; x[b,d,:] stride 1024 floats
  {
    const float* xb = x + (size_t)b * 65536 + nlo;
#pragma unroll
    for (int p = 0; p < 4; ++p) {
      int d = (tid >> 4) + p * 16;
      int r = tid & 15;
      xt[r * XT_LD + d] = xb[(size_t)d * 1024 + r];
    }
  }
  __syncthreads();

  // ---- GEMM1: logits = w * (2*X.C^T - ||c||^2)  (row-constant term dropped)
  v8f acc[8];
  v16h A0, A1;                     // A fragments, K = 0..31 / 32..63
  {
    const float* xr = &xt[ln * XT_LD];
#pragma unroll
    for (int j = 0; j < 8; ++j) {
      A0[j]     = (_Float16)xr[16 * h + j];
      A0[8 + j] = (_Float16)xr[16 * h + 8 + j];
      A1[j]     = (_Float16)xr[32 + 16 * h + j];
      A1[8 + j] = (_Float16)xr[32 + 16 * h + 8 + j];
    }
  }
#pragma unroll
  for (int t = 0; t < 8; ++t) {
    int c = c0 + t * 16 + ln;
    v16h B0 = *(const v16h*)(cb16 + (size_t)c * 64 + 16 * h);        // 32B contig
    v16h B1 = *(const v16h*)(cb16 + (size_t)c * 64 + 32 + 16 * h);
    v8f ca = {};
    ca = __builtin_amdgcn_wmma_f32_16x16x32_f16(false, A0, false, B0, (short)0, ca, false, false);
    ca = __builtin_amdgcn_wmma_f32_16x16x32_f16(false, A1, false, B1, (short)0, ca, false, false);
    float sq = sqc[c];
#pragma unroll
    for (int v = 0; v < 8; ++v) ca[v] = w * (2.0f * ca[v] - sq);
    acc[t] = ca;
  }

  // ---- wave-local softmax stats for plain logits (rows m = v + 8h)
#pragma unroll
  for (int v = 0; v < 8; ++v) {
    float m = acc[0][v];
#pragma unroll
    for (int t = 1; t < 8; ++t) m = fmaxf(m, acc[t][v]);
    for (int msk = 1; msk <= 8; msk <<= 1) m = fmaxf(m, __shfl_xor(m, msk, 32));
    float s1 = 0.f, s2 = 0.f;
#pragma unroll
    for (int t = 0; t < 8; ++t) {
      float u = acc[t][v] - m;
      float e = __expf(u);
      s1 += e; s2 += u * e;
    }
    for (int msk = 1; msk <= 8; msk <<= 1) {
      s1 += __shfl_xor(s1, msk, 32);
      s2 += __shfl_xor(s2, msk, 32);
    }
    if (ln == 0) { pmx[wave][v + 8 * h] = m; ps1[wave][v + 8 * h] = s1; ps2[wave][v + 8 * h] = s2; }
  }

  // ---- gumbel ready? (only this wave's own LDS chunk is consumed)
#if __has_builtin(__builtin_amdgcn_s_wait_asynccnt)
  __builtin_amdgcn_s_wait_asynccnt(0);
#else
  asm volatile("s_wait_asynccnt 0x0" ::: "memory");
#endif

  // ---- add gumbel, /T (T=0.5), in place; then stats for relaxed softmax
  {
    const float* gl = &glds[wave * GL_WV];
#pragma unroll
    for (int t = 0; t < 8; ++t)
#pragma unroll
      for (int v = 0; v < 8; ++v) {
        float g = gl[(v + 8 * h) * GL_LD + t * 16 + ln];
        acc[t][v] = (acc[t][v] + g) * 2.0f;
      }
  }
#pragma unroll
  for (int v = 0; v < 8; ++v) {
    float m = acc[0][v];
#pragma unroll
    for (int t = 1; t < 8; ++t) m = fmaxf(m, acc[t][v]);
    for (int msk = 1; msk <= 8; msk <<= 1) m = fmaxf(m, __shfl_xor(m, msk, 32));
    float z = 0.f;
#pragma unroll
    for (int t = 0; t < 8; ++t) z += __expf(acc[t][v] - m);
    for (int msk = 1; msk <= 8; msk <<= 1) z += __shfl_xor(z, msk, 32);
    if (ln == 0) { pgm[wave][v + 8 * h] = m; pgz[wave][v + 8 * h] = z; }
  }
  __syncthreads();

  // ---- cross-wave flash-softmax combine (16 rows, one thread each)
  if (tid < 16) {
    int r = tid;
    float M = pmx[0][r];
    for (int wv = 1; wv < 8; ++wv) M = fmaxf(M, pmx[wv][r]);
    float s1 = 0.f, s2 = 0.f;
    for (int wv = 0; wv < 8; ++wv) {
      float dm = pmx[wv][r] - M, e = __expf(dm);
      s1 += ps1[wv][r] * e;
      s2 += (ps2[wv][r] + dm * ps1[wv][r]) * e;
    }
    kldrow[r] = s2 / s1 - __logf(s1);          // sum_k p*log p for this row
    float Mg = pgm[0][r];
    for (int wv = 1; wv < 8; ++wv) Mg = fmaxf(Mg, pgm[wv][r]);
    float Zg = 0.f;
    for (int wv = 0; wv < 8; ++wv) Zg += pgz[wv][r] * __expf(pgm[wv][r] - Mg);
    rowMg[r] = Mg;
    rowIZ[r] = 1.0f / Zg;
  }
  __syncthreads();

  // ---- encodings -> f16 LDS (row major, padded)
  {
    float mg[8], iz[8];
#pragma unroll
    for (int v = 0; v < 8; ++v) { mg[v] = rowMg[v + 8 * h]; iz[v] = rowIZ[v + 8 * h]; }
#pragma unroll
    for (int t = 0; t < 8; ++t)
#pragma unroll
      for (int v = 0; v < 8; ++v) {
        float e = __expf(acc[t][v] - mg[v]) * iz[v];
        enc[(v + 8 * h) * ENC_LD + c0 + t * 16 + ln] = (_Float16)e;
      }
  }
  __syncthreads();

  // ---- GEMM2: quantized = enc @ C ; wave pairs split K (512 each)
  const int dt = wave & 3;         // d-tile 0..3
  const int kh = wave >> 2;        // K half
  const int d0 = dt * 16;
  v8f q = {};
#pragma unroll
  for (int ks = 0; ks < 16; ++ks) {
    int kb = kh * 512 + ks * 32;
    const _Float16* er = &enc[ln * ENC_LD + kb];
    v8h a0 = *(const v8h*)(er + 16 * h);
    v8h a1 = *(const v8h*)(er + 16 * h + 8);
    v16h A;
#pragma unroll
    for (int j = 0; j < 8; ++j) { A[j] = a0[j]; A[8 + j] = a1[j]; }
    v16h B = *(const v16h*)(cbT16 + (size_t)(d0 + ln) * 1024 + kb + 16 * h);
    q = __builtin_amdgcn_wmma_f32_16x16x32_f16(false, A, false, B, (short)0, q, false, false);
  }
  if (kh == 1) {
#pragma unroll
    for (int v = 0; v < 8; ++v) qpart[v + 8 * h][d0 + ln] = q[v];
  }
  __syncthreads();
  if (kh == 0) {
#pragma unroll
    for (int v = 0; v < 8; ++v) qfin[v + 8 * h][d0 + ln] = q[v] + qpart[v + 8 * h][d0 + ln];
  }
  __syncthreads();

  // ---- write quantized (back to [bs,dim,W,H]) + continuous KL partial
  float lc = 0.f;
  {
    float* ob = out_q + (size_t)b * 65536 + nlo;
    int r = tid & 15;
    int db = (tid >> 4) * 4;
#pragma unroll
    for (int i = 0; i < 4; ++i) {
      int d = db + i;
      float qv = qfin[r][d];
      ob[(size_t)d * 1024 + r] = qv;
      float df = xt[r * XT_LD + d] - qv;
      lc += df * df;
    }
  }
  red[tid] = lc;
  __syncthreads();
  for (int s = 128; s > 0; s >>= 1) {
    if (tid < s) red[tid] += red[tid + s];
    __syncthreads();
  }
  if (tid == 0) {
    float kd = 0.f;
    for (int r = 0; r < 16; ++r) kd += kldrow[r];
    partials[blockIdx.x] = w * red[0] + kd;     // /bs applied in reduce
  }
}

// ---------------------------------------------------------------------------
// Deterministic fixed-tree reduction of per-block KL partials
// ---------------------------------------------------------------------------
__global__ void vq_reduce(const float* __restrict__ partials,
                          float* __restrict__ out_scalar) {
  __shared__ float s[1024];
  int t = threadIdx.x;
  float v = 0.f;
  for (int i = t; i < NBLK; i += 1024) v += partials[i];
  s[t] = v;
  __syncthreads();
  for (int st = 512; st > 0; st >>= 1) {
    if (t < st) s[t] += s[t + st];
    __syncthreads();
  }
  if (t == 0) out_scalar[0] = s[0] * (1.0f / 64.0f);   // / bs
}

extern "C" void kernel_launch(void* const* d_in, const int* in_sizes, int n_in,
                              void* d_out, int out_size, void* d_ws, size_t ws_size,
                              hipStream_t stream) {
  (void)in_sizes; (void)n_in; (void)out_size; (void)ws_size;
  const float* x   = (const float*)d_in[0];
  const float* cb  = (const float*)d_in[1];
  const float* lvq = (const float*)d_in[2];
  const float* gum = (const float*)d_in[3];
  float* out = (float*)d_out;

  char* ws = (char*)d_ws;
  _Float16* cb16   = (_Float16*)ws;                    // 128 KB
  _Float16* cbT16  = (_Float16*)(ws + 131072);         // 128 KB
  float*    sqc    = (float*)(ws + 262144);            //   4 KB
  float*    parts  = (float*)(ws + 262144 + 4096);     //  16 KB

  vq_precompute<<<256, 256, 0, stream>>>(cb, cb16, cbT16, sqc);
  vq_fused<<<NBLK, 256, 0, stream>>>(x, gum, lvq, cb16, cbT16, sqc, out, parts);
  vq_reduce<<<1, 1024, 0, stream>>>(parts, out + 262144);
}
